// SelfMultiHeadAttention_29703993819973
// MI455X (gfx1250) — matrix operand order repaired and checked by reading, not compiled, not run on hardware
//
#include <hip/hip_runtime.h>
#include <hip/hip_bf16.h>
#include <hip/hip_fp16.h>

// Problem constants (match reference)
#define NN 4096
#define DD 768
#define HH 3
#define HDIM 256

typedef __attribute__((ext_vector_type(16))) _Float16 v16h;
typedef __attribute__((ext_vector_type(8)))  float    v8f;

#if __has_builtin(__builtin_amdgcn_tensor_load_to_lds) && \
    __has_builtin(__builtin_amdgcn_s_wait_tensorcnt)
#define USE_TDM 1
#endif

#ifdef USE_TDM
typedef unsigned int u32x4 __attribute__((ext_vector_type(4)));
typedef int          i32x4 __attribute__((ext_vector_type(4)));
typedef int          i32x8 __attribute__((ext_vector_type(8)));

// Issue one 2D TDM tile load (f16 elements): `rows` x `cols` tile, global row
// stride `gstride` (elements), LDS destination padded by (pad_amount_code+1)
// DWORDs every 2^(pad_interval_code+1) DWORDs (ISA 8.4 D# group1 encoding).
// Wave-level op: call from one wave; EXEC is ignored. Tracked by TENSORcnt.
__device__ __forceinline__ void tdm_load_2d_f16(const _Float16* gsrc,
                                                const _Float16* ldst, int cols,
                                                int rows, int gstride,
                                                int pad_interval_code,
                                                int pad_amount_code) {
  unsigned lds_off = (unsigned)(size_t)ldst;  // flat addr low 32 = LDS offset
  unsigned long long ga = (unsigned long long)(size_t)gsrc;
  u32x4 g0;
  g0[0] = 1u;                                   // count=1 (valid user D#)
  g0[1] = lds_off;                              // lds_addr [63:32]
  g0[2] = (unsigned)(ga & 0xffffffffu);         // global_addr[31:0]
  g0[3] = (unsigned)((ga >> 32) & 0x01ffffffu)  // global_addr[56:32]
          | (2u << 30);                         // type=2 ("image")
  i32x8 g1;
  g1[0] = (1 << 16)                             // data_size: 2 bytes
          | (1 << 20)                           // pad_enable
          | ((pad_interval_code & 7) << 22)     // pad interval (2^(c+1) DW)
          | ((pad_amount_code & 127) << 25);    // pad amount ((c+1) DW)
  g1[1] = (cols & 0xffff) << 16;                // tensor_dim0[15:0]
  g1[2] = ((unsigned)cols >> 16) | ((rows & 0xffff) << 16);  // dim0 hi, dim1 lo
  g1[3] = ((unsigned)rows >> 16) | ((cols & 0xffff) << 16);  // dim1 hi, tile_dim0
  g1[4] = rows & 0xffff;                        // tile_dim1 (tile_dim2 = 0)
  g1[5] = gstride;                              // tensor_dim0_stride[31:0]
  g1[6] = 0;                                    // stride hi, dim1_stride lo
  g1[7] = 0;
  i32x4 z4 = {0, 0, 0, 0};
  i32x8 z8 = {0, 0, 0, 0, 0, 0, 0, 0};
  // 6-arg form (clang-23 / therock-10.0 headers): groups 2/3 unused (2D tile).
  __builtin_amdgcn_tensor_load_to_lds(g0, g1, z4, z4, z8, 0);
}
#endif

struct HFrag {
  union {
    v16h h;
    unsigned u[8];
  };
};

// ---- WMMA fragment loaders (layouts per CDNA5 ISA 7.12.2) -------------------
// 16-bit A-matrix 16x32: lane L holds row m=L%16; VGPR v holds K pair at
// K = (v/4)*16 + (L/16)*8 + (v%4)*2.
__device__ __forceinline__ HFrag load_a16(const _Float16* base, int ld, int row0, int kk) {
  const int lane = threadIdx.x & 31;
  const int m = row0 + (lane & 15);
  const int half = lane >> 4;
  const _Float16* p = base + (size_t)m * ld + kk;
  HFrag f;
#pragma unroll
  for (int v = 0; v < 8; ++v) {
    int k = ((v >> 2) << 4) + (half << 3) + ((v & 3) << 1);
    f.u[v] = *reinterpret_cast<const unsigned*>(p + k);
  }
  return f;
}

// 16-bit B-matrix 32x16 from an [n][k] LDS tile: lane L holds column n=L%16;
// VGPR v holds K pair at K = (v/4)*16 + (L/16)*8 + (v%4)*2.
__device__ __forceinline__ HFrag load_b16_nk(const _Float16* base, int ld, int n0, int kk) {
  const int lane = threadIdx.x & 31;
  const int n = n0 + (lane & 15);
  const int half = lane >> 4;
  const _Float16* p = base + (size_t)n * ld + kk;
  HFrag f;
#pragma unroll
  for (int v = 0; v < 8; ++v)
    f.u[v] = *reinterpret_cast<const unsigned*>(p + ((v >> 2) << 4) + (half << 3) + ((v & 3) << 1));
  return f;
}

__device__ __forceinline__ v8f wmma_f16(const HFrag& a, const HFrag& b, v8f c) {
  return __builtin_amdgcn_wmma_f32_16x16x32_f16(false, a.h, false, b.h, (short)0, c,
                                                false, false);
}

// ---- f32 -> f16 conversion --------------------------------------------------
__global__ void cvt_f32_f16(const float* __restrict__ s, _Float16* __restrict__ d, int n) {
  int i = blockIdx.x * blockDim.x + threadIdx.x;
  int stride = gridDim.x * blockDim.x;
  for (; i < n; i += stride) d[i] = (_Float16)s[i];
}

// ---- Tiled WMMA GEMM: C[M,Nn] = A[M,K] @ W[Nn,K]^T + bias ------------------
// Block tile 128x64, 8 waves each doing 32x32 (2x2 WMMA 16x16x32 tiles).
#define BM 128
#define BN 64
#define BK 32
#define LDT 40  // padded f16 stride (64B row + 16B TDM pad)

__global__ __launch_bounds__(256) void gemm_f16_kernel(
    const _Float16* __restrict__ A, const _Float16* __restrict__ W,
    const float* __restrict__ bias, float* __restrict__ outF32,
    _Float16* __restrict__ outF16, int M, int Nn, int K) {
  __shared__ _Float16 As[BM * LDT];
  __shared__ _Float16 Bs[BN * LDT];

  const int tid = threadIdx.x;
  const int lane = tid & 31;
  const int wave = tid >> 5;
  const int wm = wave >> 1;   // 0..3
  const int wn = wave & 1;    // 0..1
  const int mBase = blockIdx.y * BM;
  const int nBase = blockIdx.x * BN;

  v8f acc[2][2];
#pragma unroll
  for (int i = 0; i < 2; ++i)
#pragma unroll
    for (int j = 0; j < 2; ++j) acc[i][j] = {};

  for (int k0 = 0; k0 < K; k0 += BK) {
    __syncthreads();
#ifdef USE_TDM
    // Tensor Data Mover fills both tiles (rows padded 64B->80B to match LDT).
    if (wave == 0) {
      tdm_load_2d_f16(&A[(size_t)mBase * K + k0], As, BK, BM, K, /*16 DW*/ 3,
                      /*4 DW*/ 3);
      tdm_load_2d_f16(&W[(size_t)nBase * K + k0], Bs, BK, BN, K, 3, 3);
      __builtin_amdgcn_s_wait_tensorcnt(0);
    }
#else
    // A tile: 128x32 f16 via 16B chunks
#pragma unroll
    for (int pass = 0; pass < 2; ++pass) {
      int c = tid + pass * 256;
      int row = c >> 2;
      int col = (c & 3) << 3;
      *reinterpret_cast<uint4*>(&As[row * LDT + col]) =
          *reinterpret_cast<const uint4*>(&A[(size_t)(mBase + row) * K + k0 + col]);
    }
    // W tile: 64x32 f16 (stored [n][k])
    {
      int row = tid >> 2;
      int col = (tid & 3) << 3;
      *reinterpret_cast<uint4*>(&Bs[row * LDT + col]) =
          *reinterpret_cast<const uint4*>(&W[(size_t)(nBase + row) * K + k0 + col]);
    }
#endif
    __syncthreads();

    HFrag a0 = load_a16(As, LDT, wm * 32, 0);
    HFrag a1 = load_a16(As, LDT, wm * 32 + 16, 0);
    HFrag b0 = load_b16_nk(Bs, LDT, wn * 32, 0);
    HFrag b1 = load_b16_nk(Bs, LDT, wn * 32 + 16, 0);
    acc[0][0] = wmma_f16(a0, b0, acc[0][0]);
    acc[0][1] = wmma_f16(a0, b1, acc[0][1]);
    acc[1][0] = wmma_f16(a1, b0, acc[1][0]);
    acc[1][1] = wmma_f16(a1, b1, acc[1][1]);
  }

  // Epilogue: C layout = lane holds n=lane%16, m = r + 8*(lane/16) per VGPR r
  const int halfq = lane >> 4;
  const int nloc = lane & 15;
#pragma unroll
  for (int i = 0; i < 2; ++i)
#pragma unroll
    for (int j = 0; j < 2; ++j) {
      int n = nBase + wn * 32 + j * 16 + nloc;
      float bv = bias[n];
#pragma unroll
      for (int r = 0; r < 8; ++r) {
        int m = mBase + wm * 32 + i * 16 + r + (halfq << 3);
        float val = acc[i][j][r] + bv;
        if (outF32) outF32[(size_t)m * Nn + n] = val;
        if (outF16) outF16[(size_t)m * Nn + n] = (_Float16)val;
      }
    }
}

// ---- Flash-style attention per head ----------------------------------------
// Block: 4 waves, 64 query rows (16 rows/wave). Loops over 64-key tiles.
// scores = leaky_relu(Q @ K^T); mask==0 -> -1000; online softmax; acc += P @ V.
#define LDK 264   // Qs/Ks padded stride (512B row + 16B TDM pad)
#define LDVt 68   // transposed V stride
#define LDP 72    // P staging stride

__global__ __launch_bounds__(128) void attn_kernel(
    const _Float16* __restrict__ q16, const _Float16* __restrict__ k16,
    const _Float16* __restrict__ v16, const int* __restrict__ mask,
    _Float16* __restrict__ o16) {
  extern __shared__ _Float16 smem[];
  _Float16* Qs = smem;                  // 64 * 264
  _Float16* Ks = Qs + 64 * LDK;         // 64 * 264
  _Float16* Vt = Ks + 64 * LDK;         // 256 * 68 (transposed: [d][key])
  _Float16* Ps = Vt + 256 * LDVt;       // 4 * 16 * 72 (per-wave P staging)

  const int tid = threadIdx.x;
  const int lane = tid & 31;
  const int wave = tid >> 5;
  const int halfq = lane >> 4;
  const int nloc = lane & 15;
  const int h = blockIdx.y;
  const int qbase = blockIdx.x * 64;
  const int dbase = h * HDIM;

  // Load Q tile (64 x 256) once
#ifdef USE_TDM
  if (wave == 0)
    tdm_load_2d_f16(&q16[(size_t)qbase * DD + dbase], Qs, HDIM, 64, DD,
                    /*128 DW*/ 6, /*4 DW*/ 3);
#else
#pragma unroll
  for (int i = 0; i < 16; ++i) {
    int c = tid + i * 128;
    int row = c >> 5;
    int col = (c & 31) << 3;
    *reinterpret_cast<uint4*>(&Qs[row * LDK + col]) =
        *reinterpret_cast<const uint4*>(&q16[(size_t)(qbase + row) * DD + dbase + col]);
  }
#endif

  v8f acc[16];
#pragma unroll
  for (int t = 0; t < 16; ++t) acc[t] = {};
  float rowmax[8], rowsum[8];
#pragma unroll
  for (int r = 0; r < 8; ++r) { rowmax[r] = -3.0e38f; rowsum[r] = 0.f; }

  for (int kv = 0; kv < NN; kv += 64) {
    __syncthreads();  // previous-iteration readers done
#ifdef USE_TDM
    // K tile via TDM (row pad 512B->528B == LDK); overlaps with V transpose.
    if (wave == 0)
      tdm_load_2d_f16(&k16[(size_t)kv * DD + dbase], Ks, HDIM, 64, DD, 6, 3);
#endif
    // V tile transposed into Vt (all waves)
#pragma unroll
    for (int i = 0; i < 16; ++i) {
      int c = tid + i * 128;
      int row = c >> 5;
      int col = (c & 31) << 3;
      union { uint4 q; _Float16 hh[8]; } vv;
      vv.q = *reinterpret_cast<const uint4*>(&v16[(size_t)(kv + row) * DD + dbase + col]);
#pragma unroll
      for (int j = 0; j < 8; ++j) Vt[(col + j) * LDVt + row] = vv.hh[j];
#ifndef USE_TDM
      *reinterpret_cast<uint4*>(&Ks[row * LDK + col]) =
          *reinterpret_cast<const uint4*>(&k16[(size_t)(kv + row) * DD + dbase + col]);
#endif
    }
#ifdef USE_TDM
    if (wave == 0) __builtin_amdgcn_s_wait_tensorcnt(0);  // Q (iter 0) + K done
#endif
    __syncthreads();

    // S = Q @ K^T : 16 rows x 64 keys (4 n-tiles, 8 K-steps over HD=256)
    v8f s[4];
#pragma unroll
    for (int nt = 0; nt < 4; ++nt) {
      s[nt] = {};
#pragma unroll
      for (int kd = 0; kd < 8; ++kd) {
        HFrag a = load_a16(Qs, LDK, wave * 16, kd * 32);
        HFrag b = load_b16_nk(Ks, LDK, nt * 16, kd * 32);
        s[nt] = wmma_f16(a, b, s[nt]);
      }
    }

    // leaky_relu + mask, per-row block max
    float bm[8];
#pragma unroll
    for (int r = 0; r < 8; ++r) bm[r] = -3.0e38f;
#pragma unroll
    for (int nt = 0; nt < 4; ++nt) {
#pragma unroll
      for (int r = 0; r < 8; ++r) {
        float v = s[nt][r];
        v = v > 0.f ? v : 0.01f * v;
        int mrow = qbase + wave * 16 + r + (halfq << 3);
        int mk = mask[(size_t)mrow * NN + kv + nt * 16 + nloc];
        if (mk == 0) v = -1000.f;
        s[nt][r] = v;
        bm[r] = fmaxf(bm[r], v);
      }
    }
    // reduce max across the 16 lanes of each half-wave
#pragma unroll
    for (int r = 0; r < 8; ++r) {
#pragma unroll
      for (int off = 8; off >= 1; off >>= 1)
        bm[r] = fmaxf(bm[r], __shfl_xor(bm[r], off, 32));
    }

    float corr[8];
#pragma unroll
    for (int r = 0; r < 8; ++r) {
      float nm = fmaxf(rowmax[r], bm[r]);
      corr[r] = __expf(rowmax[r] - nm);
      rowmax[r] = nm;
    }

    // P = exp(S - max): stage to LDS in row-major for A-fragment reload
    float bs[8];
#pragma unroll
    for (int r = 0; r < 8; ++r) bs[r] = 0.f;
    _Float16* Pw = Ps + wave * 16 * LDP;
#pragma unroll
    for (int nt = 0; nt < 4; ++nt) {
#pragma unroll
      for (int r = 0; r < 8; ++r) {
        float p = __expf(s[nt][r] - rowmax[r]);
        bs[r] += p;
        Pw[(r + (halfq << 3)) * LDP + nt * 16 + nloc] = (_Float16)p;
      }
    }
#pragma unroll
    for (int r = 0; r < 8; ++r) {
#pragma unroll
      for (int off = 8; off >= 1; off >>= 1)
        bs[r] += __shfl_xor(bs[r], off, 32);
      rowsum[r] = rowsum[r] * corr[r] + bs[r];
    }

    // rescale running accumulator
#pragma unroll
    for (int t = 0; t < 16; ++t)
#pragma unroll
      for (int r = 0; r < 8; ++r) acc[t][r] *= corr[r];

    // acc += P(16x64) @ V(64x256)  — per-wave LDS is in-order, no barrier needed
    HFrag p0 = load_a16(Pw, LDP, 0, 0);
    HFrag p1 = load_a16(Pw, LDP, 0, 32);
#pragma unroll
    for (int t = 0; t < 16; ++t) {
      HFrag b0 = load_b16_nk(Vt, LDVt, t * 16, 0);
      HFrag b1 = load_b16_nk(Vt, LDVt, t * 16, 32);
      acc[t] = wmma_f16(p0, b0, acc[t]);
      acc[t] = wmma_f16(p1, b1, acc[t]);
    }
  }

  // normalize and store head output (f16, feeds final GEMM)
#pragma unroll
  for (int r = 0; r < 8; ++r) rowsum[r] = 1.f / rowsum[r];
#pragma unroll
  for (int t = 0; t < 16; ++t) {
#pragma unroll
    for (int r = 0; r < 8; ++r) {
      int m = qbase + wave * 16 + r + (halfq << 3);
      o16[(size_t)m * DD + dbase + t * 16 + nloc] = (_Float16)(acc[t][r] * rowsum[r]);
    }
  }
}

// ---- host-side orchestration ------------------------------------------------
extern "C" void kernel_launch(void* const* d_in, const int* in_sizes, int n_in,
                              void* d_out, int out_size, void* d_ws, size_t ws_size,
                              hipStream_t stream) {
  const float* query = (const float*)d_in[0];
  const int*   mask  = (const int*)d_in[1];
  const float* Wq = (const float*)d_in[2];
  const float* bq = (const float*)d_in[3];
  const float* Wk = (const float*)d_in[4];
  const float* bk = (const float*)d_in[5];
  const float* Wv = (const float*)d_in[6];
  const float* bv = (const float*)d_in[7];
  const float* Wo = (const float*)d_in[8];
  const float* bo = (const float*)d_in[9];
  float* out = (float*)d_out;

  _Float16* ws = (_Float16*)d_ws;
  const size_t nd = (size_t)NN * DD;   // 3,145,728
  const size_t dd = (size_t)DD * DD;   // 589,824
  _Float16* qin16 = ws;
  _Float16* q16   = qin16 + nd;
  _Float16* k16   = q16 + nd;
  _Float16* v16   = k16 + nd;
  _Float16* o16   = v16 + nd;
  _Float16* Wq16  = o16 + nd;
  _Float16* Wk16  = Wq16 + dd;
  _Float16* Wv16  = Wk16 + dd;
  _Float16* Wo16  = Wv16 + dd;

  cvt_f32_f16<<<512, 256, 0, stream>>>(query, qin16, (int)nd);
  cvt_f32_f16<<<256, 256, 0, stream>>>(Wq, Wq16, (int)dd);
  cvt_f32_f16<<<256, 256, 0, stream>>>(Wk, Wk16, (int)dd);
  cvt_f32_f16<<<256, 256, 0, stream>>>(Wv, Wv16, (int)dd);
  cvt_f32_f16<<<256, 256, 0, stream>>>(Wo, Wo16, (int)dd);

  dim3 ggrid(DD / BN, NN / BM);  // (12, 32)
  gemm_f16_kernel<<<ggrid, 256, 0, stream>>>(qin16, Wq16, bq, nullptr, q16, NN, DD, DD);
  gemm_f16_kernel<<<ggrid, 256, 0, stream>>>(q16, Wk16, bk, nullptr, k16, NN, DD, DD);
  gemm_f16_kernel<<<ggrid, 256, 0, stream>>>(q16, Wv16, bv, nullptr, v16, NN, DD, DD);

  size_t attn_smem =
      (size_t)(64 * LDK + 64 * LDK + 256 * LDVt + 4 * 16 * LDP) * sizeof(_Float16);
  attn_kernel<<<dim3(NN / 64, HH), 128, attn_smem, stream>>>(q16, k16, v16, mask, o16);

  gemm_f16_kernel<<<ggrid, 256, 0, stream>>>(o16, Wo16, bo, out, nullptr, NN, DD, DD);
}
